// GumbelGraphNetwork_27504970564025
// MI455X (gfx1250) — matrix operand order repaired and checked by reading, not compiled, not run on hardware
//
#include <hip/hip_runtime.h>
#include <hip/hip_bf16.h>

typedef __attribute__((ext_vector_type(16))) _Float16 v16h;
typedef __attribute__((ext_vector_type(2)))  _Float16 v2h;
typedef __attribute__((ext_vector_type(8)))  float    v8f;

#define NB 4
#define NN 512
#define NF 64
#define HD 128
#define IC 4            // i-chunks for occupancy (deterministic 2-pass reduction)
#define ICH (NN / IC)   // 128 i's per chunk

// ---------- prep: P[b,n,h] = x[b,n] @ W_e1[:64] + b_e1 ; Q[b,n,h] = x[b,n] @ W_e1[64:] ----------
__global__ __launch_bounds__(256) void k_prep_pq(const float* __restrict__ x,
                                                 const float* __restrict__ We1,
                                                 const float* __restrict__ be1,
                                                 float* __restrict__ P,
                                                 float* __restrict__ Q) {
  int idx = blockIdx.x * 256 + threadIdx.x;
  if (idx >= NB * NN * HD) return;
  int h  = idx & (HD - 1);
  int bn = idx >> 7;
  const float* xr = x + (size_t)bn * NF;
  float p = be1[h], q = 0.f;
#pragma unroll 8
  for (int f = 0; f < NF; ++f) {
    float xv = xr[f];
    p += xv * We1[f * HD + h];            // innode part: rows 0..63
    q += xv * We1[(NF + f) * HD + h];     // outnode part: rows 64..127
  }
  P[idx] = p;
  Q[idx] = q;
}

// ---------- convert W_e2 (128x128, row-major in x out) to f16 ----------
__global__ __launch_bounds__(256) void k_cvt_we2(const float* __restrict__ W,
                                                 _Float16* __restrict__ Wh) {
  int idx = blockIdx.x * 256 + threadIdx.x;
  if (idx < HD * HD) Wh[idx] = (_Float16)W[idx];
}

// ---------- main edge GEMM: per (b, j-tile, i-chunk) ----------
// partial[ic][b][j][h] = sum_{i in chunk} adj[b,i,j] * relu( relu(P[j]+Q[i]) @ We2 + b_e2 )[h]
__global__ __launch_bounds__(256) void k_edge(const float* __restrict__ P,
                                              const float* __restrict__ Q,
                                              const _Float16* __restrict__ We2h,
                                              const float* __restrict__ be2,
                                              const float* __restrict__ adj,
                                              float* __restrict__ partial) {
  __shared__ __align__(16) float    Pb_lds[16][HD];         // 8 KB, persistent
  __shared__ __align__(16) float    Qi_lds[2][HD];          // double-buffered staged Q row
  __shared__ float                  Adj_lds[2][16];         // double-buffered adj tile
  __shared__ __align__(32) _Float16 Atile[2][4][32][16];    // double-buffered A fragments

  const int tid  = threadIdx.x;
  const int lane = tid & 31;
  const int nt   = tid >> 5;       // wave id == output column tile (8 x 16 cols)
  const int hi   = lane >> 4;
  const int lo   = lane & 15;

  int blk = blockIdx.x;
  const int ic = blk & (IC - 1); blk >>= 2;
  const int jt = blk & 31;       blk >>= 5;
  const int b  = blk;
  const int j0 = jt * 16;
  const int i0 = ic * ICH;

  const float* __restrict__ Qb   = Q   + (size_t)b * NN * HD;
  const float* __restrict__ adjb = adj + (size_t)b * NN * NN + j0;

  // stage the persistent P tile (16 x 128 f32)
  for (int t = tid; t < 16 * HD; t += 256) {
    int m = t >> 7, k = t & (HD - 1);
    Pb_lds[m][k] = P[((size_t)b * NN + j0 + m) * HD + k];
  }

  // B fragments for this wave's 16 output columns, all 4 K-steps.
  // 16-bit B 32x16 layout: lane (lo) = column N, K = (e>>3)*16 + hi*8 + (e&7) within a K=32 block.
  v16h Bf[4];
#pragma unroll
  for (int kb = 0; kb < 4; ++kb)
#pragma unroll
    for (int e = 0; e < 16; ++e)
      Bf[kb][e] = We2h[(kb * 32 + (e >> 3) * 16 + hi * 8 + (e & 7)) * HD + nt * 16 + lo];

  const float biasn = be2[nt * 16 + lo];
  v8f acc = {0.f, 0.f, 0.f, 0.f, 0.f, 0.f, 0.f, 0.f};

  // preamble: stage row i0 into buffer 0
  if (tid < HD)           Qi_lds[0][tid]       = Qb[(size_t)i0 * HD + tid];
  else if (tid < HD + 16) Adj_lds[0][tid - HD] = adjb[(size_t)i0 * NN + (tid - HD)];
  __syncthreads();

  for (int ii = 0; ii < ICH; ++ii) {
    const int buf = ii & 1;

    // ---- build A tile = relu(P[j]+Q[i]) in f16, packed k-pairs, fragment order ----
    // pair p: m = p>>6, k = 2*(p&63). Even/odd k share (kb, ln) and hit e, e+1.
    for (int p = tid; p < 16 * (HD / 2); p += 256) {
      int m = p >> 6, k = (p & 63) << 1;
      float2 pv = *(const float2*)&Pb_lds[m][k];
      float2 qv = *(const float2*)&Qi_lds[buf][k];
      float v0 = pv.x + qv.x; v0 = v0 > 0.f ? v0 : 0.f;
      float v1 = pv.y + qv.y; v1 = v1 > 0.f ? v1 : 0.f;
      int kb = k >> 5, r = k & 31;
      int e  = ((r >> 4) << 3) + (r & 7);              // even
      int ln = (((r >> 3) & 1) << 4) + m;
      *(v2h*)&Atile[buf][kb][ln][e] = (v2h){(_Float16)v0, (_Float16)v1};
    }
    __syncthreads();

    // ---- stage next row while WMMAs run ----
    if (ii + 1 < ICH) {
      const int i1 = i0 + ii + 1;
      if (tid < HD)           Qi_lds[buf ^ 1][tid]       = Qb[(size_t)i1 * HD + tid];
      else if (tid < HD + 16) Adj_lds[buf ^ 1][tid - HD] = adjb[(size_t)i1 * NN + (tid - HD)];
      else if (tid < HD + 21) {   // prefetch 2 ahead (speculative, drop on OOB)
        const size_t inext = (size_t)(i0 + ii + 2);
        if (tid < HD + 20) __builtin_prefetch(&Qb[inext * HD + (size_t)(tid - HD - 16) * 32], 0, 3);
        else               __builtin_prefetch(&adjb[inext * NN], 0, 3);
      }
    }

    // ---- 16x16 output tile via 4 chained WMMAs over K=128 ----
    v8f c = {0.f, 0.f, 0.f, 0.f, 0.f, 0.f, 0.f, 0.f};
#pragma unroll
    for (int kb = 0; kb < 4; ++kb) {
      v16h a = *(const v16h*)(&Atile[buf][kb][lane][0]);
      c = __builtin_amdgcn_wmma_f32_16x16x32_f16(false, a, false, Bf[kb],
                                                 (short)0, c, false, false);
    }

    // bias + relu + adjacency scale, accumulate (C layout: row M = r + hi*8, col N = lo)
#pragma unroll
    for (int r = 0; r < 8; ++r) {
      float av = Adj_lds[buf][r + hi * 8];
      float t2 = c[r] + biasn;
      t2 = t2 > 0.f ? t2 : 0.f;
      acc[r] += av * t2;
    }
    __syncthreads();   // A[buf]/Q[buf] consumed; next iter may refill them
  }

#pragma unroll
  for (int r = 0; r < 8; ++r) {
    int m = r + hi * 8;
    partial[(((size_t)ic * NB + b) * NN + (j0 + m)) * HD + nt * 16 + lo] = acc[r];
  }
}

// ---------- deterministic reduction over i-chunks ----------
__global__ __launch_bounds__(256) void k_reduce(const float* __restrict__ partial,
                                                float* __restrict__ agg) {
  int idx = blockIdx.x * 256 + threadIdx.x;
  if (idx >= NB * NN * HD) return;
  float s = 0.f;
#pragma unroll
  for (int c = 0; c < IC; ++c) s += partial[(size_t)c * NB * NN * HD + idx];
  agg[idx] = s;
}

// ---------- node MLP: out = relu(in @ W + b), rows = B*N, 128 -> 128 ----------
__global__ __launch_bounds__(256) void k_mlp(const float* __restrict__ in,
                                             const float* __restrict__ W,
                                             const float* __restrict__ bias,
                                             float* __restrict__ out) {
  int idx = blockIdx.x * 256 + threadIdx.x;
  if (idx >= NB * NN * HD) return;
  int h   = idx & (HD - 1);
  int row = idx >> 7;
  const float* ir = in + (size_t)row * HD;
  float a = bias[h];
#pragma unroll 8
  for (int k = 0; k < HD; ++k) a += ir[k] * W[k * HD + h];
  out[idx] = a > 0.f ? a : 0.f;
}

// ---------- output layer: concat(x, h2) @ W_out + b_out (+ optional skip) ----------
__global__ __launch_bounds__(256) void k_out(const float* __restrict__ x,
                                             const float* __restrict__ h2,
                                             const float* __restrict__ Wout,
                                             const float* __restrict__ bout,
                                             const int* __restrict__ skip,
                                             float* __restrict__ out) {
  int idx = blockIdx.x * 256 + threadIdx.x;
  if (idx >= NB * NN * NF) return;
  int f  = idx & (NF - 1);
  int bn = idx >> 6;
  const float* xr = x  + (size_t)bn * NF;
  const float* hr = h2 + (size_t)bn * HD;
  float a = bout[f];
#pragma unroll 8
  for (int k = 0; k < NF; ++k) a += xr[k] * Wout[k * NF + f];
#pragma unroll 8
  for (int k = 0; k < HD; ++k) a += hr[k] * Wout[(NF + k) * NF + f];
  if (*skip == 1) a += xr[f];
  out[idx] = a;
}

extern "C" void kernel_launch(void* const* d_in, const int* in_sizes, int n_in,
                              void* d_out, int out_size, void* d_ws, size_t ws_size,
                              hipStream_t stream) {
  const float* x    = (const float*)d_in[0];
  const float* adj  = (const float*)d_in[1];
  const int*   skip = (const int*)  d_in[2];
  const float* We1  = (const float*)d_in[3];
  const float* be1  = (const float*)d_in[4];
  const float* We2  = (const float*)d_in[5];
  const float* be2  = (const float*)d_in[6];
  const float* Wn1  = (const float*)d_in[7];
  const float* bn1  = (const float*)d_in[8];
  const float* Wn2  = (const float*)d_in[9];
  const float* bn2  = (const float*)d_in[10];
  const float* Wout = (const float*)d_in[11];
  const float* bout = (const float*)d_in[12];
  float* out = (float*)d_out;

  const size_t BNH = (size_t)NB * NN * HD;   // 262144
  char* ws = (char*)d_ws;
  float* P       = (float*)ws; ws += BNH * 4;
  float* Q       = (float*)ws; ws += BNH * 4;
  float* agg     = (float*)ws; ws += BNH * 4;
  float* h1      = (float*)ws; ws += BNH * 4;
  float* h2      = (float*)ws; ws += BNH * 4;
  float* partial = (float*)ws; ws += (size_t)IC * BNH * 4;
  _Float16* We2h = (_Float16*)ws; ws += (size_t)HD * HD * 2;

  int nB = (int)((BNH + 255) / 256);
  k_prep_pq<<<nB, 256, 0, stream>>>(x, We1, be1, P, Q);
  k_cvt_we2<<<(HD * HD + 255) / 256, 256, 0, stream>>>(We2, We2h);
  k_edge<<<NB * 32 * IC, 256, 0, stream>>>(P, Q, We2h, be2, adj, partial);
  k_reduce<<<nB, 256, 0, stream>>>(partial, agg);
  k_mlp<<<nB, 256, 0, stream>>>(agg, Wn1, bn1, h1);
  k_mlp<<<nB, 256, 0, stream>>>(h1, Wn2, bn2, h2);
  k_out<<<(NB * NN * NF + 255) / 256, 256, 0, stream>>>(x, h2, Wout, bout, skip, out);
}
